// RSSMCell_9380208574928
// MI455X (gfx1250) — matrix operand order repaired and checked by reading, not compiled, not run on hardware
//
#include <hip/hip_runtime.h>
#include <hip/hip_bf16.h>

// ---------------------------------------------------------------------------
// RSSM cell for MI455X (gfx1250): all GEMMs via v_wmma_f32_16x16x32_bf16,
// double-buffered LDS pipeline (1 barrier / K-step), bf16 staging via
// global_load_b128 + v_cvt_pk_bf16_f32 + ds_store_b128.
// B=8192, H=512, A=6, E=1536, Z=1024 (NCAT=32, NCLS=32)
// ---------------------------------------------------------------------------

typedef __attribute__((ext_vector_type(16))) __bf16 v16bf;
typedef __attribute__((ext_vector_type(8)))  __bf16 v8bf;
typedef __attribute__((ext_vector_type(8)))  float  v8f;

#define BM 64
#define BN 128
#define BK 32
#define LDS_STRIDE 40   // 32 + 8 pad (80B row stride -> spread across banks)

// Load 8 contiguous fp32 (8B-aligned) and convert to bf16.
__device__ __forceinline__ void load8_cvt(const float* __restrict__ src,
                                          __bf16* dst)
{
    #pragma unroll
    for (int j = 0; j < 4; ++j) {
        const float2 f = *(const float2*)(src + 2 * j);
        dst[2 * j]     = (__bf16)f.x;
        dst[2 * j + 1] = (__bf16)f.y;
    }
}

// C[M,N] = act( A[M,K] @ W[N,K]^T + bias[N] )
// A is a two-segment row-major source: cols [0,K0) from A0 (stride lda0),
// cols [K0,K) from A1 (stride lda1). K zero-padded up to a multiple of 32.
// act: 0 = identity, 1 = ELU (alpha=1) -- uniform runtime branch.
// SEG: false -> single segment, K % 32 == 0 (fully branch-free staging).
template <bool SEG>
__global__ __launch_bounds__(256) void gemm_bf16_wmma(
    const float* __restrict__ A0, int lda0, int K0,
    const float* __restrict__ A1, int lda1,
    const float* __restrict__ W,  const float* __restrict__ bias,
    float* __restrict__ C, int M, int N, int K, int act)
{
    __shared__ __bf16 lsA[2][BM][LDS_STRIDE];
    __shared__ __bf16 lsB[2][BN][LDS_STRIDE];

    const int tid    = threadIdx.x;
    const int rowBlk = blockIdx.x * BM;
    const int colBlk = blockIdx.y * BN;

    const int wave = tid >> 5;        // 0..7
    const int lane = tid & 31;
    const int wm   = wave >> 2;       // 0..1  (row of 32x32 wave tiles)
    const int wn   = wave & 3;        // 0..3  (col of 32x32 wave tiles)
    const int hlf  = lane >> 4;       // 0..1
    const int l16  = lane & 15;

    // Staging coordinates (hoisted out of the K loop)
    const int  arow  = tid >> 2;               // A: 4 threads per row
    const int  acs   = (tid & 3) * 8;          // A: 8-elem chunk
    const long agrow = rowBlk + arow;
    const int  brow  = tid >> 1;               // W: 2 threads per row
    const int  bcs   = (tid & 1) * 16;         // W: 16-elem chunk
    const long bgrow = colBlk + brow;

    // --- staging helpers (global fp32 -> bf16 regs) ---
    auto loadA = [&](int kbase, __bf16* tmp) {
        const int kc = kbase + acs;
        if (!SEG || kc + 8 <= K0) {
            load8_cvt(A0 + agrow * (long)lda0 + kc, tmp);
        } else if (kc >= K0 && kc + 8 <= K) {
            load8_cvt(A1 + agrow * (long)lda1 + (kc - K0), tmp);
        } else {   // boundary / tail chunk (rare)
            #pragma unroll
            for (int j = 0; j < 8; ++j) {
                const int k = kc + j;
                float v = 0.f;
                if (k < K0)      v = A0[agrow * (long)lda0 + k];
                else if (k < K)  v = A1[agrow * (long)lda1 + (k - K0)];
                tmp[j] = (__bf16)v;
            }
        }
    };
    auto loadB = [&](int kbase, __bf16* tmp) {
        const int kc = kbase + bcs;
        if (!SEG || kc + 16 <= K) {
            load8_cvt(W + bgrow * (long)K + kc,     tmp);
            load8_cvt(W + bgrow * (long)K + kc + 8, tmp + 8);
        } else {   // tail chunk with zero-fill (rare)
            #pragma unroll
            for (int j = 0; j < 16; ++j) {
                const int k = kc + j;
                tmp[j] = (__bf16)((k < K) ? W[bgrow * (long)K + k] : 0.f);
            }
        }
    };

    v8f acc[2][2];
    #pragma unroll
    for (int i = 0; i < 2; ++i)
        #pragma unroll
        for (int j = 0; j < 2; ++j) {
            v8f zv = {0.f,0.f,0.f,0.f,0.f,0.f,0.f,0.f};
            acc[i][j] = zv;
        }

    const int ksteps = (K + BK - 1) / BK;

    // --- prologue: stage tile 0 into buffer 0 ---
    __bf16 atmp[8], btmp[16];
    loadA(0, atmp);
    loadB(0, btmp);
    *(v8bf*)&lsA[0][arow][acs]      = *(v8bf*)atmp;
    *(v8bf*)&lsB[0][brow][bcs]      = *(v8bf*)btmp;
    *(v8bf*)&lsB[0][brow][bcs + 8]  = *(v8bf*)(btmp + 8);
    __syncthreads();

    int buf = 0;
    for (int ks = 0; ks < ksteps; ++ks) {
        const bool has_next = (ks + 1 < ksteps);   // uniform scalar branch
        if (has_next) {                            // prefetch tile ks+1 -> regs
            loadA((ks + 1) * BK, atmp);
            loadB((ks + 1) * BK, btmp);
        }

        // --- 2x2 WMMA tiles per wave from buffer `buf` ---
        v16bf bfrag[2];
        #pragma unroll
        for (int ni = 0; ni < 2; ++ni) {
            const int bc = wn * 32 + ni * 16 + l16;  // B col (lane = col)
            const int kb = hlf * 8;                  // K = {kb..kb+7, kb+16..kb+23}
            v8bf blo = *(const v8bf*)&lsB[buf][bc][kb];
            v8bf bhi = *(const v8bf*)&lsB[buf][bc][kb + 16];
            bfrag[ni] = __builtin_shufflevector(blo, bhi,
                0,1,2,3,4,5,6,7,8,9,10,11,12,13,14,15);
        }
        #pragma unroll
        for (int mi = 0; mi < 2; ++mi) {
            const int ar = wm * 32 + mi * 16 + l16;  // A row (lane = row)
            const int ka = hlf * 8;                  // K = {ka..ka+7, ka+16..ka+23}
            v8bf alo = *(const v8bf*)&lsA[buf][ar][ka];
            v8bf ahi = *(const v8bf*)&lsA[buf][ar][ka + 16];
            v16bf afrag = __builtin_shufflevector(alo, ahi,
                0,1,2,3,4,5,6,7,8,9,10,11,12,13,14,15);
            #pragma unroll
            for (int ni = 0; ni < 2; ++ni) {
                acc[mi][ni] = __builtin_amdgcn_wmma_f32_16x16x32_bf16(
                    false, afrag, false, bfrag[ni], (short)0, acc[mi][ni],
                    false, false);
            }
        }

        if (has_next) {                            // store prefetch -> other buffer
            *(v8bf*)&lsA[buf ^ 1][arow][acs]     = *(v8bf*)atmp;
            *(v8bf*)&lsB[buf ^ 1][brow][bcs]     = *(v8bf*)btmp;
            *(v8bf*)&lsB[buf ^ 1][brow][bcs + 8] = *(v8bf*)(btmp + 8);
        }
        __syncthreads();
        buf ^= 1;
    }

    // --- epilogue: bias (+ELU), scatter per C-layout (M = v + hlf*8, N = l16)
    #pragma unroll
    for (int mi = 0; mi < 2; ++mi) {
        #pragma unroll
        for (int ni = 0; ni < 2; ++ni) {
            const int n  = colBlk + wn * 32 + ni * 16 + l16;
            const float bv = bias[n];
            #pragma unroll
            for (int v = 0; v < 8; ++v) {
                const int m = rowBlk + wm * 32 + mi * 16 + v + hlf * 8;
                float val = acc[mi][ni][v] + bv;
                if (act == 1) val = (val > 0.f) ? val : (expf(val) - 1.f);
                C[(long)m * N + n] = val;
            }
        }
    }
}

// --- GRU gate fusion: h = (1-u)*n + u*h_prev ---
__global__ void gru_gate_kernel(const float* __restrict__ gi,
                                const float* __restrict__ gh,
                                const float* __restrict__ h_prev,
                                float* __restrict__ h_out,
                                int B, int H)
{
    const int i = blockIdx.x * blockDim.x + threadIdx.x;
    if (i >= B * H) return;
    const int b = i / H;
    const int j = i - b * H;
    const float* gib = gi + (long)b * 3 * H;
    const float* ghb = gh + (long)b * 3 * H;
    const float r = 1.f / (1.f + expf(-(gib[j]         + ghb[j])));
    const float u = 1.f / (1.f + expf(-(gib[H + j]     + ghb[H + j])));
    const float n = tanhf(gib[2 * H + j] + r * ghb[2 * H + j]);
    h_out[i] = (1.f - u) * n + u * h_prev[i];
}

// --- per-(b,cat) softmax + gumbel-argmax, one wave32 per category ---
__global__ void sampler_kernel(const float* __restrict__ post_logits,
                               const float* __restrict__ gumbel,
                               float* __restrict__ z, int ncats_total)
{
    const int gwarp = (blockIdx.x * blockDim.x + threadIdx.x) >> 5;
    const int lane  = threadIdx.x & 31;
    if (gwarp >= ncats_total) return;

    const long base = (long)gwarp * 32;
    const float v = post_logits[base + lane];
    const float g = gumbel[base + lane];

    // softmax over 32 classes (wave32 reductions)
    float m = v;
    #pragma unroll
    for (int off = 16; off > 0; off >>= 1)
        m = fmaxf(m, __shfl_xor(m, off, 32));
    const float e = expf(v - m);
    float s = e;
    #pragma unroll
    for (int off = 16; off > 0; off >>= 1)
        s += __shfl_xor(s, off, 32);
    const float prob = e / s;

    // argmax(v + gumbel), first-index tie-break (matches jnp.argmax)
    float key = v + g;
    int   idx = lane;
    #pragma unroll
    for (int off = 16; off > 0; off >>= 1) {
        const float ok = __shfl_xor(key, off, 32);
        const int   oi = __shfl_xor(idx, off, 32);
        if (ok > key || (ok == key && oi < idx)) { key = ok; idx = oi; }
    }
    const float onehot = (lane == idx) ? 1.f : 0.f;
    z[base + lane] = (onehot + prob) - prob;   // straight-through forward value
}

extern "C" void kernel_launch(void* const* d_in, const int* in_sizes, int n_in,
                              void* d_out, int out_size, void* d_ws, size_t ws_size,
                              hipStream_t stream)
{
    (void)in_sizes; (void)n_in; (void)out_size; (void)ws_size;

    constexpr int B = 8192, H = 512, Aa = 6, E = 1536;
    constexpr int NCAT = 32, NCLS = 32, Z = NCAT * NCLS;   // 1024
    constexpr int H3 = 3 * H;                               // 1536

    const float* h_prev = (const float*)d_in[0];
    const float* z_prev = (const float*)d_in[1];
    const float* action = (const float*)d_in[2];
    const float* embed  = (const float*)d_in[3];
    const float* gumbel = (const float*)d_in[4];
    const float* W_ih   = (const float*)d_in[5];
    const float* b_ih   = (const float*)d_in[6];
    const float* W_hh   = (const float*)d_in[7];
    const float* b_hh   = (const float*)d_in[8];
    const float* pw1    = (const float*)d_in[9];
    const float* pb1    = (const float*)d_in[10];
    const float* pw2    = (const float*)d_in[11];
    const float* pb2    = (const float*)d_in[12];
    const float* qw1    = (const float*)d_in[13];
    const float* qb1    = (const float*)d_in[14];
    const float* qw2    = (const float*)d_in[15];
    const float* qb2    = (const float*)d_in[16];

    // d_out regions (return order: h, z, prior_r, post_r)
    float* out_h     = (float*)d_out;
    float* out_z     = out_h + (long)B * H;
    float* out_prior = out_z + (long)B * Z;
    float* out_post  = out_prior + (long)B * Z;

    // workspace: gi[B,3H] | gh[B,3H] | hid1[B,H] | hid2[B,H]
    char* ws = (char*)d_ws;
    float* gi   = (float*)ws;                                  ws += (size_t)B * H3 * 4;
    float* gh   = (float*)ws;                                  ws += (size_t)B * H3 * 4;
    float* hid1 = (float*)ws;                                  ws += (size_t)B * H  * 4;
    float* hid2 = (float*)ws;

    const dim3 blk(256);

    // 1) gi = concat(z_prev, action) @ W_ih^T + b_ih   [B, 3H], K = Z + A = 1030
    gemm_bf16_wmma<true><<<dim3(B / BM, H3 / BN), blk, 0, stream>>>(
        z_prev, Z, Z, action, Aa, W_ih, b_ih, gi, B, H3, Z + Aa, 0);

    // 2) gh = h_prev @ W_hh^T + b_hh   [B, 3H], K = H
    gemm_bf16_wmma<false><<<dim3(B / BM, H3 / BN), blk, 0, stream>>>(
        h_prev, H, H, h_prev, H, W_hh, b_hh, gh, B, H3, H, 0);

    // 3) GRU gates -> h (directly into d_out)
    gru_gate_kernel<<<(B * H + 255) / 256, blk, 0, stream>>>(gi, gh, h_prev, out_h, B, H);

    // 4) hid1 = ELU(h @ pw1^T + pb1)   [B, H]
    gemm_bf16_wmma<false><<<dim3(B / BM, H / BN), blk, 0, stream>>>(
        out_h, H, H, out_h, H, pw1, pb1, hid1, B, H, H, 1);

    // 5) prior_logits = hid1 @ pw2^T + pb2   [B, Z]  (straight into d_out)
    gemm_bf16_wmma<false><<<dim3(B / BM, Z / BN), blk, 0, stream>>>(
        hid1, H, H, hid1, H, pw2, pb2, out_prior, B, Z, H, 0);

    // 6) hid2 = ELU(concat(h, embed) @ qw1^T + qb1)   [B, H], K = H + E = 2048
    gemm_bf16_wmma<true><<<dim3(B / BM, H / BN), blk, 0, stream>>>(
        out_h, H, H, embed, E, qw1, qb1, hid2, B, H, H + E, 1);

    // 7) post_logits = hid2 @ qw2^T + qb2   [B, Z]  (straight into d_out)
    gemm_bf16_wmma<false><<<dim3(B / BM, Z / BN), blk, 0, stream>>>(
        hid2, H, H, hid2, H, qw2, qb2, out_post, B, Z, H, 0);

    // 8) softmax + gumbel argmax + straight-through z (one wave per (b,cat))
    const int ncats_total = B * NCAT;
    sampler_kernel<<<(ncats_total * 32 + 255) / 256, blk, 0, stream>>>(
        out_post, gumbel, out_z, ncats_total);
}
